// MultiHeadAttention_60361470378485
// MI455X (gfx1250) — matrix operand order repaired and checked by reading, not compiled, not run on hardware
//
#include <hip/hip_runtime.h>
#include <stdint.h>

// MHA forward B=2,T=2048,C=1024,H=16,D=64 on gfx1250 (MI455X).
// GEMMs: 64x64 wave tiles through v_wmma_f32_16x16x32_bf16, __launch_bounds__
// to avoid accumulator spills. Attention: flash, K/V staged to LDS via TDM
// (tensor_load_to_lds), double buffered, s_wait_tensorcnt + barriers.

#define B_ 2
#define T_ 2048
#define C_ 1024
#define H_ 16
#define D_ 64

typedef __bf16 bf16;
typedef __attribute__((ext_vector_type(16))) __bf16 v16bf;
typedef __attribute__((ext_vector_type(8)))  __bf16 v8bf;
typedef __attribute__((ext_vector_type(8)))  float  v8f;
typedef __attribute__((ext_vector_type(4)))  unsigned int v4u;
typedef __attribute__((ext_vector_type(8)))  int v8i;
typedef __attribute__((ext_vector_type(4)))  int v4i;

static __device__ __forceinline__ v8f wmma_bf16(v16bf a, v16bf b, v8f c) {
  return __builtin_amdgcn_wmma_f32_16x16x32_bf16(false, a, false, b, (short)0, c,
                                                 false, false);
}

// A-fragment: 16x32 bf16 from row-major [rows][K]; ISA 16-bit A layout (8+8 K split)
static __device__ __forceinline__ v16bf load_frag_a(const bf16* base, int row0,
                                                    int k0, int ld) {
  int lane = threadIdx.x & 31;
  const bf16* p = base + (size_t)(row0 + (lane & 15)) * ld + k0 + (lane >> 4) * 8;
  union { v16bf v; v8bf h[2]; } u;
  u.h[0] = *(const v8bf*)(p);
  u.h[1] = *(const v8bf*)(p + 16);
  return u.v;
}

// B-fragment: 32x16 bf16 sourced from row-major B^T [N][K]; ISA 16-bit layout (8+8 K split)
static __device__ __forceinline__ v16bf load_frag_b(const bf16* base, int n0,
                                                    int k0, int ld) {
  int lane = threadIdx.x & 31;
  const bf16* p = base + (size_t)(n0 + (lane & 15)) * ld + k0 + (lane >> 4) * 8;
  union { v16bf v; v8bf h[2]; } u;
  u.h[0] = *(const v8bf*)(p);
  u.h[1] = *(const v8bf*)(p + 16);
  return u.v;
}

static __device__ __forceinline__ unsigned lds_off(const void* p) {
  return (unsigned)(unsigned long long)(uintptr_t)p;   // low 32 bits = LDS byte addr
}

// TDM 2D tile load: copy tile_d1 rows x tile_d0 elements (bf16) from global
// (row stride stride0 elements) into packed LDS [tile_d1][tile_d0].
static __device__ __forceinline__ void tdm_load_2d(unsigned lds_addr,
                                                   const bf16* gptr,
                                                   unsigned tile_d0, unsigned tile_d1,
                                                   unsigned tens_d0, unsigned tens_d1,
                                                   unsigned stride0) {
  unsigned long long ga = (unsigned long long)(uintptr_t)gptr;
  v4u g0;
  g0[0] = 1u;                                            // count=1 (user descriptor)
  g0[1] = lds_addr;                                      // lds_addr
  g0[2] = (unsigned)(ga & 0xffffffffu);                  // global_addr[31:0]
  g0[3] = (unsigned)((ga >> 32) & 0x1ffffffu) | 0x80000000u;  // [56:32] | type=2
  v8i g1;
  g1[0] = (int)(1u << 16);                               // data_size=1 -> 2 bytes
  g1[1] = (int)((tens_d0 & 0xffffu) << 16);              // tensor_dim0[15:0]
  g1[2] = (int)(((tens_d0 >> 16) & 0xffffu) | ((tens_d1 & 0xffffu) << 16));
  g1[3] = (int)(((tens_d1 >> 16) & 0xffffu) | ((tile_d0 & 0xffffu) << 16));
  g1[4] = (int)(tile_d1 & 0xffffu);                      // tile_dim1, tile_dim2=0
  g1[5] = (int)stride0;                                  // tensor_dim0_stride[31:0]
  g1[6] = 0;
  g1[7] = 0;
  v4i z4 = {0, 0, 0, 0};
  v8i z8 = {0, 0, 0, 0, 0, 0, 0, 0};
  __builtin_amdgcn_tensor_load_to_lds(g0, g1, z4, z4, z8, 0);
}

// ---------------- conversion kernels ----------------
__global__ void mha_cvt_f32_bf16(const float* __restrict__ src,
                                 bf16* __restrict__ dst, int n) {
  int i = blockIdx.x * blockDim.x + threadIdx.x;
  if (i < n) dst[i] = (bf16)src[i];
}

// W [H][C][D] f32 -> Wt [H][D][C] bf16
__global__ void mha_cvt_w_transpose(const float* __restrict__ w,
                                    bf16* __restrict__ wt) {
  int i = blockIdx.x * blockDim.x + threadIdx.x;
  if (i >= H_ * C_ * D_) return;
  int d = i % D_;
  int c = (i / D_) % C_;
  int h = i / (C_ * D_);
  wt[((size_t)h * D_ + d) * C_ + c] = (bf16)w[i];
}

// ---------------- QKV projections: one wave = 64(row) x 64(=D) tile ----------------
__global__ void __launch_bounds__(128, 1)
mha_qkv_gemm(const bf16* __restrict__ xb,
             const bf16* __restrict__ wqt,
             const bf16* __restrict__ wkt,
             const bf16* __restrict__ wvt,
             bf16* __restrict__ q, bf16* __restrict__ k,
             bf16* __restrict__ vT) {
  int gw   = blockIdx.x * (blockDim.x >> 5) + (threadIdx.x >> 5);
  int tt   = gw & (T_ / 64 - 1);          // 0..31  64-row tile
  int bh   = (gw >> 5) & (B_ * H_ - 1);   // 0..31
  int kind = gw >> 10;                    // 0=q 1=k 2=v
  int b = bh >> 4, h = bh & (H_ - 1);
  const bf16* xrow = xb + ((size_t)b * T_ + tt * 64) * C_;
  const bf16* wt = (kind == 0 ? wqt : (kind == 1 ? wkt : wvt)) + (size_t)h * D_ * C_;

  v8f acc[4][4] = {};
  for (int kc = 0; kc < C_; kc += 32) {
    v16bf a[4], bb[4];
#pragma unroll
    for (int g = 0; g < 4; ++g) a[g] = load_frag_a(xrow, g * 16, kc, C_);
#pragma unroll
    for (int j = 0; j < 4; ++j) bb[j] = load_frag_b(wt, j * 16, kc, C_);
#pragma unroll
    for (int g = 0; g < 4; ++g)
#pragma unroll
      for (int j = 0; j < 4; ++j)
        acc[g][j] = wmma_bf16(a[g], bb[j], acc[g][j]);
  }

  int lane = threadIdx.x & 31, n = lane & 15, hi = lane >> 4;
  if (kind == 2) {                 // V stored transposed: [bh][D][T]
    bf16* out = vT + (size_t)bh * D_ * T_;
#pragma unroll
    for (int g = 0; g < 4; ++g)
#pragma unroll
      for (int j = 0; j < 4; ++j)
#pragma unroll
        for (int r = 0; r < 8; ++r)
          out[(size_t)(j * 16 + n) * T_ + tt * 64 + g * 16 + r + 8 * hi] =
              (bf16)acc[g][j][r];
  } else {                          // row-major [bh][T][D]
    bf16* out = (kind == 0 ? q : k) + (size_t)bh * T_ * D_;
#pragma unroll
    for (int g = 0; g < 4; ++g)
#pragma unroll
      for (int j = 0; j < 4; ++j)
#pragma unroll
        for (int r = 0; r < 8; ++r)
          out[(size_t)(tt * 64 + g * 16 + r + 8 * hi) * D_ + j * 16 + n] =
              (bf16)acc[g][j][r];
  }
}

// ---------------- causal flash attention ----------------
// Block = 4 waves = 4 adjacent 16-row Q tiles of one (b,h).
// K/V 32-key blocks staged into LDS by wave0 via TDM, double buffered.
__global__ void __launch_bounds__(128, 1)
mha_attn_fwd(const bf16* __restrict__ q,
             const bf16* __restrict__ k,
             const bf16* __restrict__ vT,
             bf16* __restrict__ o) {
  int wave = threadIdx.x >> 5;
  int lane = threadIdx.x & 31;
  int n = lane & 15, hi = lane >> 4;
  int bh = blockIdx.x >> 5;            // 0..31
  int qg = blockIdx.x & 31;            // Q group: 4 tiles
  int q0 = (qg * 4 + wave) * 16;       // this wave's Q rows
  int q0max = (qg * 4 + 3) * 16;       // last wave's

  const bf16* qh = q  + (size_t)bh * T_ * D_;
  const bf16* kh = k  + (size_t)bh * T_ * D_;
  const bf16* vh = vT + (size_t)bh * D_ * T_;

  __shared__ __align__(16) bf16 kbuf[2][32 * 64];   // [keys 32][D 64]
  __shared__ __align__(16) bf16 vbuf[2][64 * 32];   // [D 64][keys 32]
  __shared__ __align__(16) bf16 pbuf[4][16 * 32];
  bf16* pl = pbuf[wave];

  // prime stage 0
  if (wave == 0) {
    tdm_load_2d(lds_off(&kbuf[0][0]), kh, 64, 32, 64, T_, 64);
    tdm_load_2d(lds_off(&vbuf[0][0]), vh, 32, 64, T_, 64, T_);
  }

  v16bf qa0 = load_frag_a(qh, q0, 0, D_);
  v16bf qa1 = load_frag_a(qh, q0, 32, D_);

  v8f acc[4] = {};
  float mrow[8], lrow[8];
#pragma unroll
  for (int r = 0; r < 8; ++r) { mrow[r] = -__builtin_inff(); lrow[r] = 0.f; }

  const float sc = 0.03125f;  // 1/sqrt(C)  (reference scales by C, not D)

  int it = 0;
  for (int kb = 0; kb <= q0max; kb += 32, ++it) {
    int cur = it & 1, nxt = cur ^ 1;
    if (wave == 0) __builtin_amdgcn_s_wait_tensorcnt(0);   // stage `cur` landed
    __syncthreads();
    if (wave == 0 && kb + 32 <= q0max) {                   // start stage `nxt`
      tdm_load_2d(lds_off(&kbuf[nxt][0]), kh + (size_t)(kb + 32) * D_,
                  64, 32, 64, T_, 64);
      tdm_load_2d(lds_off(&vbuf[nxt][0]), vh + (kb + 32),
                  32, 64, T_, 64, T_);
    }

    if (kb <= q0) {   // this wave still has work in this key block (wave-uniform)
      const bf16* kc_ = kbuf[cur];
      const bf16* vb_ = vbuf[cur];
      v8f s0 = {}, s1 = {};
      s0 = wmma_bf16(qa0, load_frag_b(kc_, 0,  0,  64), s0);
      s0 = wmma_bf16(qa1, load_frag_b(kc_, 0,  32, 64), s0);
      s1 = wmma_bf16(qa0, load_frag_b(kc_, 16, 0,  64), s1);
      s1 = wmma_bf16(qa1, load_frag_b(kc_, 16, 32, 64), s1);

      bool need_mask = (kb + 31 > q0);
      float p0[8], p1[8];
#pragma unroll
      for (int r = 0; r < 8; ++r) {
        int row = q0 + r + 8 * hi;
        float x0 = s0[r] * sc;
        float x1 = s1[r] * sc;
        if (need_mask) {
          if (kb + n > row)      x0 = -__builtin_inff();
          if (kb + 16 + n > row) x1 = -__builtin_inff();
        }
        float mx = fmaxf(x0, x1);
#pragma unroll
        for (int off = 8; off >= 1; off >>= 1)
          mx = fmaxf(mx, __shfl_xor(mx, off, 16));
        float mnew = fmaxf(mrow[r], mx);
        float corr = __expf(mrow[r] - mnew);
        float e0 = __expf(x0 - mnew);
        float e1 = __expf(x1 - mnew);
        float ls = e0 + e1;
#pragma unroll
        for (int off = 8; off >= 1; off >>= 1)
          ls += __shfl_xor(ls, off, 16);
        lrow[r] = lrow[r] * corr + ls;
        mrow[r] = mnew;
#pragma unroll
        for (int j = 0; j < 4; ++j) acc[j][r] *= corr;
        p0[r] = e0;
        p1[r] = e1;
      }

      // reshape P (C-layout) -> A-fragment layout through LDS
#pragma unroll
      for (int r = 0; r < 8; ++r) {
        pl[(r + 8 * hi) * 32 + n]      = (bf16)p0[r];
        pl[(r + 8 * hi) * 32 + 16 + n] = (bf16)p1[r];
      }
      asm volatile("s_wait_dscnt 0" ::: "memory");
      v16bf pa = load_frag_a(pl, 0, 0, 32);
#pragma unroll
      for (int j = 0; j < 4; ++j)
        acc[j] = wmma_bf16(pa, load_frag_b(vb_, j * 16, 0, 32), acc[j]);
    }
    __syncthreads();   // everyone done with stage `cur` before it is overwritten
  }

  // normalize + write concat-head layout [B][T][H*D]
  int bq = bh >> 4, hq = bh & (H_ - 1);
#pragma unroll
  for (int r = 0; r < 8; ++r) {
    float inv = 1.0f / lrow[r];
    int t = q0 + r + 8 * hi;
    size_t rowoff = ((size_t)bq * T_ + t) * C_ + (size_t)hq * D_;
#pragma unroll
    for (int j = 0; j < 4; ++j)
      o[rowoff + j * 16 + n] = (bf16)(acc[j][r] * inv);
  }
}

// ---------------- output projection: one wave = 64x64 tile of O @ Wo^T + bo ------
__global__ void __launch_bounds__(128, 1)
mha_out_proj(const bf16* __restrict__ o,
             const bf16* __restrict__ wo,
             const float* __restrict__ bo,
             float* __restrict__ out) {
  int gw = blockIdx.x * (blockDim.x >> 5) + (threadIdx.x >> 5);
  int nt = gw & (C_ / 64 - 1);   // 0..15
  int mt = gw >> 4;              // 0..63  (B*T/64)

  v8f acc[4][4] = {};
  for (int kc = 0; kc < C_; kc += 32) {
    v16bf a[4], bb[4];
#pragma unroll
    for (int g = 0; g < 4; ++g) a[g] = load_frag_a(o, mt * 64 + g * 16, kc, C_);
#pragma unroll
    for (int j = 0; j < 4; ++j)
      bb[j] = load_frag_b(wo, nt * 64 + j * 16, kc, C_);
#pragma unroll
    for (int g = 0; g < 4; ++g)
#pragma unroll
      for (int j = 0; j < 4; ++j)
        acc[g][j] = wmma_bf16(a[g], bb[j], acc[g][j]);
  }

  int lane = threadIdx.x & 31, n = lane & 15, hi = lane >> 4;
#pragma unroll
  for (int j = 0; j < 4; ++j) {
    int col = nt * 64 + j * 16 + n;
    float bias = bo[col];
#pragma unroll
    for (int g = 0; g < 4; ++g)
#pragma unroll
      for (int r = 0; r < 8; ++r)
        out[(size_t)(mt * 64 + g * 16 + r + 8 * hi) * C_ + col] =
            acc[g][j][r] + bias;
  }
}

extern "C" void kernel_launch(void* const* d_in, const int* in_sizes, int n_in,
                              void* d_out, int out_size, void* d_ws, size_t ws_size,
                              hipStream_t stream) {
  const float* x  = (const float*)d_in[0];
  const float* Wq = (const float*)d_in[1];
  const float* Wk = (const float*)d_in[2];
  const float* Wv = (const float*)d_in[3];
  const float* Wo = (const float*)d_in[4];
  const float* bo = (const float*)d_in[5];
  float* out = (float*)d_out;

  char* ws = (char*)d_ws;
  size_t off = 0;
  auto walloc = [&](size_t bytes) -> char* {
    char* p = ws + off;
    off += (bytes + 255) & ~(size_t)255;
    return p;
  };
  bf16* xb   = (bf16*)walloc((size_t)B_ * T_ * C_ * 2);
  bf16* wqt  = (bf16*)walloc((size_t)H_ * D_ * C_ * 2);
  bf16* wkt  = (bf16*)walloc((size_t)H_ * D_ * C_ * 2);
  bf16* wvt  = (bf16*)walloc((size_t)H_ * D_ * C_ * 2);
  bf16* wob  = (bf16*)walloc((size_t)C_ * C_ * 2);
  bf16* qws  = (bf16*)walloc((size_t)B_ * H_ * T_ * D_ * 2);
  bf16* kws  = (bf16*)walloc((size_t)B_ * H_ * T_ * D_ * 2);
  bf16* vtws = (bf16*)walloc((size_t)B_ * H_ * D_ * T_ * 2);  // V^T
  bf16* ob   = (bf16*)walloc((size_t)B_ * T_ * C_ * 2);

  const int thr = 256;
  mha_cvt_f32_bf16<<<(B_ * T_ * C_ + thr - 1) / thr, thr, 0, stream>>>(
      x, xb, B_ * T_ * C_);
  mha_cvt_f32_bf16<<<(C_ * C_ + thr - 1) / thr, thr, 0, stream>>>(Wo, wob, C_ * C_);
  mha_cvt_w_transpose<<<(H_ * C_ * D_ + thr - 1) / thr, thr, 0, stream>>>(Wq, wqt);
  mha_cvt_w_transpose<<<(H_ * C_ * D_ + thr - 1) / thr, thr, 0, stream>>>(Wk, wkt);
  mha_cvt_w_transpose<<<(H_ * C_ * D_ + thr - 1) / thr, thr, 0, stream>>>(Wv, wvt);

  // 3 kinds * 32 (b,h) * 32 64-row tiles = 3072 waves / 4 per block
  mha_qkv_gemm<<<3 * B_ * H_ * (T_ / 64) / 4, 128, 0, stream>>>(xb, wqt, wkt, wvt,
                                                                qws, kws, vtws);
  // 32 (b,h) * 32 Q-groups (4 tiles each) = 1024 blocks
  mha_attn_fwd<<<B_ * H_ * (T_ / 64), 128, 0, stream>>>(qws, kws, vtws, ob);
  // 64 M-tiles * 16 N-tiles = 1024 waves / 4 per block
  mha_out_proj<<<(B_ * T_ / 64) * (C_ / 64) / 4, 128, 0, stream>>>(ob, wob, bo, out);
}
